// Seq2SeqLSTM_v0_32856499814911
// MI455X (gfx1250) — matrix operand order repaired and checked by reading, not compiled, not run on hardware
//
#include <hip/hip_runtime.h>

typedef __attribute__((ext_vector_type(16))) _Float16 v16h;
typedef __attribute__((ext_vector_type(8)))  float    v8f;
typedef unsigned int u32;
typedef __attribute__((ext_vector_type(4))) u32 u32x4;
typedef __attribute__((ext_vector_type(8))) int i32x8;
typedef __attribute__((ext_vector_type(4))) int i32x4;

#define TSTEPS 12
#define NNODES 50000
#define DIN    32
#define HID    128
#define DEC    7
#define MT     4             // M-tiles per workgroup
#define NB     (MT*16)       // 64 nodes per workgroup
#define KT_CAT 8             // 256 / 32
#define NT_CAT 32            // 512 / 16

// f16 workspace layout (element = half)
#define WS_ENCB  0                               // 8 tiles  * 512
#define WS_WE    (WS_ENCB + 8*512)               // 8*32 tiles * 512
#define WS_WD    (WS_WE + KT_CAT*NT_CAT*512)
#define WS_FC1   (WS_WD + KT_CAT*NT_CAT*512)     // 4*4 tiles * 512
#define WS_TOTAL (WS_FC1 + 16*512)

// ---------------------------------------------------------------------------
// Prep: weights -> f16 in exact WMMA B-operand swizzled tile layout.
// B element (k,n) of a 32x16 tile -> lane = 16*((k>>3)&1)+n, half = (k&7)+8*(k>>4).
// ---------------------------------------------------------------------------
__global__ void prep_weights(const float* __restrict__ encW,
                             const float* __restrict__ Wih_e, const float* __restrict__ Whh_e,
                             const float* __restrict__ Wih_d, const float* __restrict__ Whh_d,
                             const float* __restrict__ fc1W,
                             _Float16* __restrict__ ws) {
  int idx = blockIdx.x * blockDim.x + threadIdx.x;
  if (idx >= WS_TOTAL) return;
  int r    = idx & 511;
  int lane = r >> 4;
  int hh   = r & 15;
  int k    = (hh < 8) ? (8 * (lane >> 4) + hh) : (16 + 8 * (lane >> 4) + (hh - 8));
  int n    = lane & 15;
  float v;
  if (idx < WS_WE) {
    int nt = idx >> 9;
    int j  = nt * 16 + n;
    v = encW[j * DIN + k];
  } else if (idx < WS_WD) {
    int t  = (idx - WS_WE) >> 9;
    int kt = t >> 5, nt = t & 31;
    int K  = kt * 32 + k, j = nt * 16 + n;
    v = (K < HID) ? Wih_e[j * HID + K] : Whh_e[j * HID + (K - HID)];
  } else if (idx < WS_FC1) {
    int t  = (idx - WS_WD) >> 9;
    int kt = t >> 5, nt = t & 31;
    int K  = kt * 32 + k, j = nt * 16 + n;
    v = (K < HID) ? Wih_d[j * HID + K] : Whh_d[j * HID + (K - HID)];
  } else {
    int t  = (idx - WS_FC1) >> 9;
    int kt = t >> 2, nt = t & 3;
    int K  = kt * 32 + k, j = nt * 16 + n;
    v = fc1W[j * HID + K];
  }
  ws[idx] = (_Float16)v;
}

// ---------------------------------------------------------------------------
// TDM: 1-D contiguous global->LDS copy (nbytes multiple of 8, n8 < 65536).
// 6-arg builtin form (clang-23 / therock headers).
// ---------------------------------------------------------------------------
__device__ __forceinline__ void tdm_load_1d(const void* gsrc, unsigned lds_off, unsigned nbytes) {
  unsigned n8 = nbytes >> 3;                       // 8-byte elements
  unsigned long long ga = (unsigned long long)gsrc;
  u32x4 g0;
  g0[0] = 1u;                                      // count=1, user descriptor
  g0[1] = lds_off;                                 // lds_addr (bytes)
  g0[2] = (u32)(ga & 0xFFFFFFFFu);                 // global_addr[31:0]
  g0[3] = (u32)((ga >> 32) & 0x1FFFFFFu) | (2u << 30);  // global_addr[56:32] | type=2
  i32x8 g1;
  g1[0] = (int)(3u << 16);                         // wg_mask=0, data_size=3 (8B)
  g1[1] = (int)((n8 & 0xFFFFu) << 16);             // tensor_dim0[15:0]
  g1[2] = (int)(((n8 >> 16) & 0xFFFFu) | (1u << 16));   // tensor_dim0[31:16], tensor_dim1=1
  g1[3] = (int)((n8 & 0xFFFFu) << 16);             // tile_dim0 = n8
  g1[4] = 0;                                       // tile_dim1/2 unused
  g1[5] = (int)n8;                                 // tensor_dim0_stride low
  g1[6] = 0;
  g1[7] = 0;
  i32x4 z4;
  z4[0] = 0; z4[1] = 0; z4[2] = 0; z4[3] = 0;
  i32x8 z8;
  z8[0] = 0; z8[1] = 0; z8[2] = 0; z8[3] = 0;
  z8[4] = 0; z8[5] = 0; z8[6] = 0; z8[7] = 0;
  __builtin_amdgcn_tensor_load_to_lds(g0, g1, z4, z4, z8, 0);
}

__device__ __forceinline__ float sigf(float x) { return 1.0f / (1.0f + __expf(-x)); }
__device__ __forceinline__ float tanh_fast(float x) {
  x = fminf(fmaxf(x, -12.0f), 12.0f);
  float e = __expf(2.0f * x);
  return (e - 1.0f) / (e + 1.0f);
}

// ---------------------------------------------------------------------------
// Fused seq2seq LSTM: one workgroup = 64 nodes (4 M-tiles), 8 waves; each
// wave owns 16 hidden columns; c-state lives in accumulator registers.
// ---------------------------------------------------------------------------
__global__ __launch_bounds__(256, 1)
void lstm_fused(const float* __restrict__ x,
                const float* __restrict__ enc_b,
                const float* __restrict__ bih_e, const float* __restrict__ bhh_e,
                const float* __restrict__ bih_d, const float* __restrict__ bhh_d,
                const float* __restrict__ fc1_b,
                const float* __restrict__ fc2_W, const float* __restrict__ fc2_b,
                const _Float16* __restrict__ ws,
                float* __restrict__ out) {
  __shared__ __align__(32) _Float16 sWB[KT_CAT * NT_CAT * 512];  // 256 KB gate weights
  __shared__ __align__(32) _Float16 sAux[16 * 512];              // encB (8 KB) then fc1 (16 KB)
  __shared__ __align__(32) _Float16 sA[MT * 8 * 512];            // A tiles per M-tile: kt0-3 X, kt4-7 h
  __shared__ float sBias[512];
  __shared__ float sEncB[HID];
  __shared__ float sF1B[64];
  __shared__ float sF2W[64];
  __shared__ float sF2B;
  __shared__ float sPredPart[4 * NB];

  const int tid     = threadIdx.x;
  const int wv      = tid >> 5;
  const int lane    = tid & 31;
  const int node0   = blockIdx.x * NB;
  const int colLane = lane & 15;
  const int mHi     = lane >> 4;
  const int jcol    = wv * 16 + colLane;

  // ---- encoder-phase LDS fill: big buffers via TDM, small via VALU ----
  if (wv == 0) {
    tdm_load_1d(ws + WS_WE, (u32)(unsigned long long)(uintptr_t)sWB, KT_CAT * NT_CAT * 512 * 2);
    tdm_load_1d(ws + WS_ENCB, (u32)(unsigned long long)(uintptr_t)sAux, 8 * 512 * 2);
    __builtin_amdgcn_s_wait_tensorcnt(0);
  } else {
    for (int i = tid - 32; i < 512; i += 224) sBias[i] = bih_e[i] + bhh_e[i];
    for (int i = tid - 32; i < HID; i += 224) sEncB[i] = enc_b[i];
    for (int i = tid - 32; i < MT * 8 * 512; i += 224) sA[i] = (_Float16)0.0f;  // h0 = 0
  }
  __syncthreads();

  v8f cacc[MT];
  #pragma unroll
  for (int mt = 0; mt < MT; ++mt)
    #pragma unroll
    for (int e = 0; e < 8; ++e) cacc[mt][e] = 0.0f;

  // A-layout scatter coords for owned column jcol
  const int kk_in = jcol & 31;
  const int sc_al = 16 * ((kk_in >> 3) & 1);
  const int sc_ah = (kk_in & 7) + 8 * (kk_in >> 4);
  const int mrow  = colLane;
  const int dlo   = mHi ? 8 : 0;

  // ======================= encoder =======================
  for (int t = 0; t < TSTEPS; ++t) {
    // --- X_t = relu(x_t @ encW^T + enc_b) -> sA kt0..3 per M-tile ---
    {
      v16h bw = *(const v16h*)(&sAux[wv * 512 + lane * 16]);
      float bv = sEncB[jcol];
      #pragma unroll
      for (int mt = 0; mt < MT; ++mt) {
        int row = node0 + mt * 16 + mrow;
        row = row < NNODES ? row : NNODES - 1;
        const float4* xr4 = (const float4*)(x + (size_t)t * NNODES * DIN + (size_t)row * DIN);
        if (t + 1 < TSTEPS)
          __builtin_prefetch(x + (size_t)(t + 1) * NNODES * DIN + (size_t)row * DIN, 0, 0);
        float xv[16];
        float4 p;
        p = xr4[(dlo >> 2) + 0]; xv[0] = p.x;  xv[1] = p.y;  xv[2] = p.z;  xv[3] = p.w;
        p = xr4[(dlo >> 2) + 1]; xv[4] = p.x;  xv[5] = p.y;  xv[6] = p.z;  xv[7] = p.w;
        p = xr4[(dlo >> 2) + 4]; xv[8] = p.x;  xv[9] = p.y;  xv[10] = p.z; xv[11] = p.w;
        p = xr4[(dlo >> 2) + 5]; xv[12] = p.x; xv[13] = p.y; xv[14] = p.z; xv[15] = p.w;
        v16h ax;
        #pragma unroll
        for (int q = 0; q < 16; ++q) ax[q] = (_Float16)xv[q];
        v8f acc;
        #pragma unroll
        for (int e = 0; e < 8; ++e) acc[e] = bv;
        acc = __builtin_amdgcn_wmma_f32_16x16x32_f16(false, ax, false, bw, (short)0, acc, false, false);
        _Float16* base = &sA[(mt * 8 + (jcol >> 5)) * 512];
        #pragma unroll
        for (int e = 0; e < 8; ++e) {
          int m = e + 8 * mHi;
          base[(sc_al + m) * 16 + sc_ah] = (_Float16)fmaxf(acc[e], 0.0f);
        }
      }
    }
    __syncthreads();

    // --- Z[64,512] = [X|H][64,256] @ Wcat[256,512]; B amortized over 4 M-tiles ---
    v8f z[4 * MT];
    #pragma unroll
    for (int g = 0; g < 4; ++g) {
      float bv = sBias[g * HID + jcol];
      #pragma unroll
      for (int mt = 0; mt < MT; ++mt)
        #pragma unroll
        for (int e = 0; e < 8; ++e) z[g * MT + mt][e] = bv;
    }
    #pragma unroll
    for (int kt = 0; kt < KT_CAT; ++kt) {
      v16h a[MT];
      #pragma unroll
      for (int mt = 0; mt < MT; ++mt)
        a[mt] = *(const v16h*)(&sA[(mt * 8 + kt) * 512 + lane * 16]);
      #pragma unroll
      for (int g = 0; g < 4; ++g) {
        v16h b = *(const v16h*)(&sWB[(kt * NT_CAT + g * 8 + wv) * 512 + lane * 16]);
        #pragma unroll
        for (int mt = 0; mt < MT; ++mt)
          z[g * MT + mt] = __builtin_amdgcn_wmma_f32_16x16x32_f16(false, a[mt], false, b, (short)0, z[g * MT + mt], false, false);
      }
    }
    __syncthreads();

    // --- LSTM cell nonlinearity, h -> sA kt4..7 ---
    #pragma unroll
    for (int mt = 0; mt < MT; ++mt) {
      _Float16* base = &sA[(mt * 8 + 4 + (jcol >> 5)) * 512];
      #pragma unroll
      for (int e = 0; e < 8; ++e) {
        float iv = sigf(z[0 * MT + mt][e]);
        float fv = sigf(z[1 * MT + mt][e]);
        float gv = tanh_fast(z[2 * MT + mt][e]);
        float ov = sigf(z[3 * MT + mt][e]);
        float c  = fv * cacc[mt][e] + iv * gv;
        cacc[mt][e] = c;
        float h  = ov * tanh_fast(c);
        int m = e + 8 * mHi;
        base[(sc_al + m) * 16 + sc_ah] = (_Float16)h;
      }
    }
    __syncthreads();
  }

  // ============== phase switch: decoder input = h_enc (fixed) ==============
  for (int i = tid; i < MT * 4 * 512; i += 256) {
    int mt = i / (4 * 512);
    int r  = i - mt * (4 * 512);
    sA[mt * 8 * 512 + r] = sA[mt * 8 * 512 + 4 * 512 + r];
  }
  if (wv == 0) {
    tdm_load_1d(ws + WS_WD, (u32)(unsigned long long)(uintptr_t)sWB, KT_CAT * NT_CAT * 512 * 2);
    tdm_load_1d(ws + WS_FC1, (u32)(unsigned long long)(uintptr_t)sAux, 16 * 512 * 2);
    __builtin_amdgcn_s_wait_tensorcnt(0);
  } else {
    for (int i = tid - 32; i < 512; i += 224) sBias[i] = bih_d[i] + bhh_d[i];
    for (int i = tid - 32; i < 64; i += 224) { sF1B[i] = fc1_b[i]; sF2W[i] = fc2_W[i]; }
    if (tid == 32) sF2B = fc2_b[0];
  }
  __syncthreads();

  // ======================= decoder =======================
  for (int s = 0; s < DEC; ++s) {
    v8f z[4 * MT];
    #pragma unroll
    for (int g = 0; g < 4; ++g) {
      float bv = sBias[g * HID + jcol];
      #pragma unroll
      for (int mt = 0; mt < MT; ++mt)
        #pragma unroll
        for (int e = 0; e < 8; ++e) z[g * MT + mt][e] = bv;
    }
    #pragma unroll
    for (int kt = 0; kt < KT_CAT; ++kt) {
      v16h a[MT];
      #pragma unroll
      for (int mt = 0; mt < MT; ++mt)
        a[mt] = *(const v16h*)(&sA[(mt * 8 + kt) * 512 + lane * 16]);
      #pragma unroll
      for (int g = 0; g < 4; ++g) {
        v16h b = *(const v16h*)(&sWB[(kt * NT_CAT + g * 8 + wv) * 512 + lane * 16]);
        #pragma unroll
        for (int mt = 0; mt < MT; ++mt)
          z[g * MT + mt] = __builtin_amdgcn_wmma_f32_16x16x32_f16(false, a[mt], false, b, (short)0, z[g * MT + mt], false, false);
      }
    }
    __syncthreads();

    #pragma unroll
    for (int mt = 0; mt < MT; ++mt) {
      _Float16* base = &sA[(mt * 8 + 4 + (jcol >> 5)) * 512];
      #pragma unroll
      for (int e = 0; e < 8; ++e) {
        float iv = sigf(z[0 * MT + mt][e]);
        float fv = sigf(z[1 * MT + mt][e]);
        float gv = tanh_fast(z[2 * MT + mt][e]);
        float ov = sigf(z[3 * MT + mt][e]);
        float c  = fv * cacc[mt][e] + iv * gv;
        cacc[mt][e] = c;
        float h  = ov * tanh_fast(c);
        int m = e + 8 * mHi;
        base[(sc_al + m) * 16 + sc_ah] = (_Float16)h;
      }
    }
    __syncthreads();

    // --- head: relu(h @ fc1^T + b1) @ fc2^T + b2, waves 0..3 ---
    if (wv < 4) {
      v8f u[MT];
      float bv = sF1B[jcol];
      #pragma unroll
      for (int mt = 0; mt < MT; ++mt)
        #pragma unroll
        for (int e = 0; e < 8; ++e) u[mt][e] = bv;
      #pragma unroll
      for (int kk = 0; kk < 4; ++kk) {
        v16h b = *(const v16h*)(&sAux[(kk * 4 + wv) * 512 + lane * 16]);
        #pragma unroll
        for (int mt = 0; mt < MT; ++mt) {
          v16h a = *(const v16h*)(&sA[(mt * 8 + 4 + kk) * 512 + lane * 16]);
          u[mt] = __builtin_amdgcn_wmma_f32_16x16x32_f16(false, a, false, b, (short)0, u[mt], false, false);
        }
      }
      float wj = sF2W[jcol];
      #pragma unroll
      for (int mt = 0; mt < MT; ++mt) {
        float part[8];
        #pragma unroll
        for (int e = 0; e < 8; ++e) part[e] = wj * fmaxf(u[mt][e], 0.0f);
        #pragma unroll
        for (int off = 1; off < 16; off <<= 1) {
          #pragma unroll
          for (int e = 0; e < 8; ++e) part[e] += __shfl_xor(part[e], off, 32);
        }
        if (colLane == 0) {
          #pragma unroll
          for (int e = 0; e < 8; ++e) sPredPart[wv * NB + mt * 16 + e + 8 * mHi] = part[e];
        }
      }
    }
    __syncthreads();
    if (tid < NB) {
      int node = node0 + tid;
      if (node < NNODES) {
        float p = sF2B + sPredPart[tid] + sPredPart[NB + tid] +
                  sPredPart[2 * NB + tid] + sPredPart[3 * NB + tid];
        out[(size_t)node * DEC + s] = p;
      }
    }
  }
}

// ---------------------------------------------------------------------------
extern "C" void kernel_launch(void* const* d_in, const int* in_sizes, int n_in,
                              void* d_out, int out_size, void* d_ws, size_t ws_size,
                              hipStream_t stream) {
  (void)in_sizes; (void)n_in; (void)out_size; (void)ws_size;
  const float* x     = (const float*)d_in[0];
  const float* encW  = (const float*)d_in[1];
  const float* enc_b = (const float*)d_in[2];
  const float* Wih_e = (const float*)d_in[3];
  const float* Whh_e = (const float*)d_in[4];
  const float* bih_e = (const float*)d_in[5];
  const float* bhh_e = (const float*)d_in[6];
  const float* Wih_d = (const float*)d_in[7];
  const float* Whh_d = (const float*)d_in[8];
  const float* bih_d = (const float*)d_in[9];
  const float* bhh_d = (const float*)d_in[10];
  const float* fc1W  = (const float*)d_in[11];
  const float* fc1b  = (const float*)d_in[12];
  const float* fc2W  = (const float*)d_in[13];
  const float* fc2b  = (const float*)d_in[14];
  _Float16* ws = (_Float16*)d_ws;

  prep_weights<<<(WS_TOTAL + 255) / 256, 256, 0, stream>>>(encW, Wih_e, Whh_e, Wih_d, Whh_d, fc1W, ws);
  lstm_fused<<<(NNODES + NB - 1) / NB, 256, 0, stream>>>(x, enc_b, bih_e, bhh_e, bih_d, bhh_d,
                                                         fc1b, fc2W, fc2b, ws, (float*)d_out);
}
